// SASRec_14362370638462
// MI455X (gfx1250) — matrix operand order repaired and checked by reading, not compiled, not run on hardware
//
#include <hip/hip_runtime.h>
#include <hip/hip_bf16.h>

typedef __attribute__((ext_vector_type(16))) __bf16 bf16x16;
typedef __attribute__((ext_vector_type(8)))  float  f32x8;

#define NEGBIG (-1.0e9f)

// ---------------------------------------------------------------- utilities
__global__ void f32_to_bf16_kernel(const float* __restrict__ x,
                                   __bf16* __restrict__ y, int n) {
  for (int i = blockIdx.x * blockDim.x + threadIdx.x; i < n;
       i += gridDim.x * blockDim.x)
    y[i] = (__bf16)x[i];
}

// ------------------------------------------------------------------- embed
// seqs = (item_emb[id]*sqrt(D) + pos_emb[l]) * (id != 0)
__global__ __launch_bounds__(128) void embed_kernel(
    const int* __restrict__ logs, const float* __restrict__ item,
    const float* __restrict__ pos, float* __restrict__ sf,
    __bf16* __restrict__ sb, int L) {
  int m = blockIdx.x;            // token index (b*L + l)
  int t = threadIdx.x;           // feature
  int id = logs[m];
  float v = item[(size_t)id * 128 + t] * 11.3137084989847604f +
            pos[(size_t)(m % L) * 128 + t];
  if (id == 0) v = 0.f;
  sf[(size_t)m * 128 + t] = v;
  sb[(size_t)m * 128 + t] = (__bf16)v;
}

// --------------------------------------------------------------- layernorm
__global__ __launch_bounds__(128) void ln_kernel(
    const float* __restrict__ x, const float* __restrict__ s,
    const float* __restrict__ b, float* __restrict__ yf,
    __bf16* __restrict__ yb) {
  __shared__ float red[128];
  int m = blockIdx.x, t = threadIdx.x;
  float v = x[(size_t)m * 128 + t];
  red[t] = v;
  __syncthreads();
  for (int off = 64; off > 0; off >>= 1) {
    if (t < off) red[t] += red[t + off];
    __syncthreads();
  }
  float mu = red[0] * (1.f / 128.f);
  __syncthreads();
  float d = v - mu;
  red[t] = d * d;
  __syncthreads();
  for (int off = 64; off > 0; off >>= 1) {
    if (t < off) red[t] += red[t + off];
    __syncthreads();
  }
  float var = red[0] * (1.f / 128.f);
  float y = d * rsqrtf(var + 1e-8f) * s[t] + b[t];
  if (yf) yf[(size_t)m * 128 + t] = y;
  if (yb) yb[(size_t)m * 128 + t] = (__bf16)y;
}

// ------------------------------------------------------------ generic GEMM
// Y(MxN) = X(MxK,bf16) @ W^T (W is NxK,bf16) + bias, optional relu,
// +residual(MxN,f32), token mask. One wave per 32x64 tile (2 M-tiles reuse B).
__global__ __launch_bounds__(128) void gemm_bf16_kernel(
    const __bf16* __restrict__ X, const __bf16* __restrict__ W,
    const float* __restrict__ bias, const float* __restrict__ residual,
    const int* __restrict__ mask_ids, float* __restrict__ Yf,
    __bf16* __restrict__ Yb, int M, int N, int K, int relu) {
  int wave   = blockIdx.x * (blockDim.x >> 5) + (threadIdx.x >> 5);
  int strips = N >> 6;
  int tileM2 = wave / strips;
  int nb0    = (wave % strips) << 6;
  if (tileM2 * 32 >= M) return;
  int lane = threadIdx.x & 31;
  int half = lane >> 4;
  int l16  = lane & 15;
  size_t rowA0 = (size_t)tileM2 * 32 + l16;
  size_t rowA1 = rowA0 + 16;

  f32x8 acc[2][4] = {};
#pragma unroll 4
  for (int k0 = 0; k0 < K; k0 += 32) {
    bf16x16 a[2];
    const __bf16* ap0 = X + rowA0 * K + k0 + half * 8;
    const __bf16* ap1 = X + rowA1 * K + k0 + half * 8;
#pragma unroll
    for (int e = 0; e < 8; ++e) {
      a[0][e] = ap0[e]; a[0][8 + e] = ap0[16 + e];
      a[1][e] = ap1[e]; a[1][8 + e] = ap1[16 + e];
    }
    bf16x16 bm[4];
#pragma unroll
    for (int j = 0; j < 4; ++j) {
      const __bf16* bp = W + (size_t)(nb0 + j * 16 + l16) * K + k0 + half * 8;
#pragma unroll
      for (int e = 0; e < 16; ++e) bm[j][e] = bp[e + (e & 8)];
    }
#pragma unroll
    for (int mi = 0; mi < 2; ++mi)
#pragma unroll
      for (int j = 0; j < 4; ++j)
        acc[mi][j] = __builtin_amdgcn_wmma_f32_16x16x32_bf16(
            false, a[mi], false, bm[j], (short)0, acc[mi][j], false, false);
  }
#pragma unroll
  for (int mi = 0; mi < 2; ++mi)
#pragma unroll
    for (int j = 0; j < 4; ++j) {
      int n = nb0 + j * 16 + l16;
      float bv = bias ? bias[n] : 0.f;
#pragma unroll
      for (int i = 0; i < 8; ++i) {
        int m = tileM2 * 32 + mi * 16 + half * 8 + i;
        float v = acc[mi][j][i] + bv;
        if (relu) v = fmaxf(v, 0.f);
        if (residual) v += residual[(size_t)m * N + n];
        if (mask_ids && mask_ids[m] == 0) v = 0.f;
        if (Yf) Yf[(size_t)m * N + n] = v;
        if (Yb) Yb[(size_t)m * N + n] = (__bf16)v;
      }
    }
}

// ----------------------------------------------------------------- KV GEMM
// KV = seqs @ [Wk;Wv]^T + b  (N=256, K=128).  K half -> Kb (M,128) row-major.
// V half -> VT[(b*2+h)*64+dh][key] (key padded to Lpad), so the attention
// P@V B-fragments become contiguous per-lane loads. The transposed store is
// contiguous per lane because consecutive acc elements hold consecutive keys.
__global__ __launch_bounds__(128) void kv_gemm_kernel(
    const __bf16* __restrict__ X, const __bf16* __restrict__ W,
    const float* __restrict__ bias, __bf16* __restrict__ Kb,
    __bf16* __restrict__ VT, int M, int K, int Lseq, int Lpad) {
  const int N = 256;
  int wave   = blockIdx.x * (blockDim.x >> 5) + (threadIdx.x >> 5);
  int tileM2 = wave >> 2;            // strips = 4
  int nb0    = (wave & 3) << 6;
  if (tileM2 * 32 >= M) return;
  int lane = threadIdx.x & 31;
  int half = lane >> 4;
  int l16  = lane & 15;
  size_t rowA0 = (size_t)tileM2 * 32 + l16;
  size_t rowA1 = rowA0 + 16;

  f32x8 acc[2][4] = {};
#pragma unroll 4
  for (int k0 = 0; k0 < K; k0 += 32) {
    bf16x16 a[2];
    const __bf16* ap0 = X + rowA0 * K + k0 + half * 8;
    const __bf16* ap1 = X + rowA1 * K + k0 + half * 8;
#pragma unroll
    for (int e = 0; e < 8; ++e) {
      a[0][e] = ap0[e]; a[0][8 + e] = ap0[16 + e];
      a[1][e] = ap1[e]; a[1][8 + e] = ap1[16 + e];
    }
    bf16x16 bm[4];
#pragma unroll
    for (int j = 0; j < 4; ++j) {
      const __bf16* bp = W + (size_t)(nb0 + j * 16 + l16) * K + k0 + half * 8;
#pragma unroll
      for (int e = 0; e < 16; ++e) bm[j][e] = bp[e + (e & 8)];
    }
#pragma unroll
    for (int mi = 0; mi < 2; ++mi)
#pragma unroll
      for (int j = 0; j < 4; ++j)
        acc[mi][j] = __builtin_amdgcn_wmma_f32_16x16x32_bf16(
            false, a[mi], false, bm[j], (short)0, acc[mi][j], false, false);
  }
#pragma unroll
  for (int mi = 0; mi < 2; ++mi)
#pragma unroll
    for (int j = 0; j < 4; ++j) {
      int n = nb0 + j * 16 + l16;
      float bv = bias[n];
      if (n < 128) {
#pragma unroll
        for (int i = 0; i < 8; ++i) {
          int m = tileM2 * 32 + mi * 16 + half * 8 + i;
          Kb[(size_t)m * 128 + n] = (__bf16)(acc[mi][j][i] + bv);
        }
      } else {
        int h = (n - 128) >> 6, dh = (n - 128) & 63;
        int m0 = tileM2 * 32 + mi * 16 + half * 8;
        int b = m0 / Lseq, l0 = m0 - b * Lseq;   // 8 keys never cross a batch
        __bf16* vp = VT + (((size_t)b * 2 + h) * 64 + dh) * Lpad + l0;
#pragma unroll
        for (int i = 0; i < 8; ++i)
          vp[i] = (__bf16)(acc[mi][j][i] + bv);
      }
    }
}

// --------------------------------------------------------------- attention
// One wave per (b, h, 16-query tile). Q bf16 (M,128) n=h*64+dh.
// Kb bf16 (M,128); VT bf16 [(b*2+h)*64+dh][Lpad].
__global__ __launch_bounds__(32) void attn_kernel(
    const __bf16* __restrict__ Qm, const __bf16* __restrict__ Kb,
    const __bf16* __restrict__ VT, __bf16* __restrict__ Om, int L) {
  __shared__ float  sS[16 * 208];
  __shared__ __bf16 sP[16 * 224];
  int nqt = (L + 15) / 16;           // 13
  int qt = blockIdx.x % nqt;
  int h  = (blockIdx.x / nqt) & 1;
  int b  = blockIdx.x / (nqt * 2);
  int lane = threadIdx.x;
  int half = lane >> 4, l16 = lane & 15;
  int q0 = qt * 16;

  // preload Q A-fragments (hd=64 -> two 16x32 k-steps)
  int qr = q0 + l16; if (qr >= L) qr = L - 1;
  const __bf16* qp = Qm + ((size_t)(b * L + qr)) * 128 + h * 64;
  bf16x16 aq[2];
#pragma unroll
  for (int kk = 0; kk < 2; ++kk) {
    int base = kk * 32 + half * 8;
#pragma unroll
    for (int e = 0; e < 8; ++e) {
      aq[kk][e] = qp[base + e];
      aq[kk][8 + e] = qp[base + 16 + e];
    }
  }
  // scores = Q K^T * 0.125 with causal mask -> LDS
  for (int kt = 0; kt < nqt; ++kt) {
    int kcol = kt * 16 + l16;
    int kc = kcol < L ? kcol : L - 1;
    const __bf16* kp = Kb + ((size_t)(b * L + kc)) * 128 + h * 64;
    bf16x16 bm[2];
#pragma unroll
    for (int kk = 0; kk < 2; ++kk) {
      int base = kk * 32 + half * 8;
#pragma unroll
      for (int e = 0; e < 16; ++e) bm[kk][e] = kp[base + e + (e & 8)];
    }
    f32x8 s{};
#pragma unroll
    for (int kk = 0; kk < 2; ++kk)
      s = __builtin_amdgcn_wmma_f32_16x16x32_bf16(
          false, aq[kk], false, bm[kk], (short)0, s, false, false);
#pragma unroll
    for (int i = 0; i < 8; ++i) {
      int qrow = half * 8 + i;
      int qg = q0 + qrow;
      int kg = kt * 16 + l16;
      float v = (kg <= qg && kg < L) ? s[i] * 0.125f : NEGBIG;
      sS[qrow * 208 + kg] = v;
    }
  }
  __syncthreads();
  // row softmax, write P as bf16 (pad K to 224 with zeros)
  for (int r = 0; r < 16; ++r) {
    float mx = -3.0e38f;
    for (int k = lane; k < 208; k += 32) mx = fmaxf(mx, sS[r * 208 + k]);
    for (int off = 16; off > 0; off >>= 1) mx = fmaxf(mx, __shfl_xor(mx, off, 32));
    float sum = 0.f;
    for (int k = lane; k < 208; k += 32) {
      float e = __expf(sS[r * 208 + k] - mx);
      sS[r * 208 + k] = e;
      sum += e;
    }
    for (int off = 16; off > 0; off >>= 1) sum += __shfl_xor(sum, off, 32);
    float inv = 1.f / sum;
    for (int k = lane; k < 224; k += 32)
      sP[r * 224 + k] = (k < 208) ? (__bf16)(sS[r * 208 + k] * inv) : (__bf16)0.f;
  }
  __syncthreads();
  // O = P @ V   (K dim padded to 224 -> 7 k-steps; 4 output col tiles)
  f32x8 o[4] = {};
  for (int kc = 0; kc < 7; ++kc) {
    bf16x16 a;
    int kb = kc * 32 + half * 8;
#pragma unroll
    for (int e = 0; e < 8; ++e) {
      a[e] = sP[l16 * 224 + kb + e];
      a[8 + e] = sP[l16 * 224 + kb + 16 + e];
    }
    int kb2 = kc * 32 + half * 8;
    bf16x16 bm[4];
#pragma unroll
    for (int j = 0; j < 4; ++j) {
      // contiguous per-lane load from transposed V (P is 0 beyond L)
      const __bf16* vp =
          VT + (((size_t)b * 2 + h) * 64 + j * 16 + l16) * 224 + kb2;
#pragma unroll
      for (int e = 0; e < 16; ++e) bm[j][e] = vp[e + (e & 8)];
    }
#pragma unroll
    for (int j = 0; j < 4; ++j)
      o[j] = __builtin_amdgcn_wmma_f32_16x16x32_bf16(
          false, a, false, bm[j], (short)0, o[j], false, false);
  }
#pragma unroll
  for (int j = 0; j < 4; ++j)
#pragma unroll
    for (int i = 0; i < 8; ++i) {
      int qrow = half * 8 + i;
      int qg = q0 + qrow;
      if (qg < L)
        Om[((size_t)(b * L + qg)) * 128 + h * 64 + j * 16 + l16] =
            (__bf16)o[j][i];
    }
}

// -------------------------------------------------- neighbor head (+= out)
__global__ __launch_bounds__(128) void nbr_kernel(
    const int* __restrict__ logs, const int* __restrict__ nbr,
    const float* __restrict__ item, const float* __restrict__ user,
    const float* __restrict__ lns, const float* __restrict__ lnb,
    float* __restrict__ out) {
  __shared__ float kn[10 * 128];
  __shared__ float red[128];
  int m = blockIdx.x, t = threadIdx.x;
  int id = logs[m];
  float q = item[(size_t)id * 128 + t];
#pragma unroll
  for (int n = 0; n < 10; ++n)
    kn[n * 128 + t] = user[(size_t)nbr[(size_t)m * 10 + n] * 128 + t];
  __syncthreads();
  float sc[10];
#pragma unroll 1
  for (int n = 0; n < 10; ++n) {
    red[t] = q * kn[n * 128 + t];
    __syncthreads();
    for (int off = 64; off > 0; off >>= 1) {
      if (t < off) red[t] += red[t + off];
      __syncthreads();
    }
    sc[n] = red[0] * 0.0883883476483184f;   // 1/sqrt(128)
    __syncthreads();
    if (nbr[(size_t)m * 10 + n] == 0) sc[n] = NEGBIG;
  }
  float mx = sc[0];
#pragma unroll
  for (int n = 1; n < 10; ++n) mx = fmaxf(mx, sc[n]);
  float sum = 0.f;
#pragma unroll
  for (int n = 0; n < 10; ++n) { sc[n] = __expf(sc[n] - mx); sum += sc[n]; }
  float inv = 1.f / sum;
  float a = q;
#pragma unroll
  for (int n = 0; n < 10; ++n) a += sc[n] * inv * kn[n * 128 + t];
  // layernorm(a)
  red[t] = a;
  __syncthreads();
  for (int off = 64; off > 0; off >>= 1) {
    if (t < off) red[t] += red[t + off];
    __syncthreads();
  }
  float mu = red[0] * (1.f / 128.f);
  __syncthreads();
  float d = a - mu;
  red[t] = d * d;
  __syncthreads();
  for (int off = 64; off > 0; off >>= 1) {
    if (t < off) red[t] += red[t + off];
    __syncthreads();
  }
  float var = red[0] * (1.f / 128.f);
  float y = d * rsqrtf(var + 1e-8f) * lns[t] + lnb[t];
  out[(size_t)m * 128 + t] += y;
}

// ------------------------------------------------------------------- host
extern "C" void kernel_launch(void* const* d_in, const int* in_sizes, int n_in,
                              void* d_out, int out_size, void* d_ws,
                              size_t ws_size, hipStream_t stream) {
  (void)in_sizes; (void)n_in; (void)out_size; (void)ws_size;
  const int Bn = 256, L = 200, D = 128, Lpad = 224;
  const int M = Bn * L;                       // 51200 tokens

  const int*   logs      = (const int*)d_in[0];
  const int*   nbr       = (const int*)d_in[1];
  const float* item_emb  = (const float*)d_in[2];
  const float* user_emb  = (const float*)d_in[3];
  const float* pos_emb   = (const float*)d_in[4];
  const float* attn_ln_s = (const float*)d_in[5];
  const float* attn_ln_b = (const float*)d_in[6];
  const float* Wqkv      = (const float*)d_in[7];
  const float* bqkv      = (const float*)d_in[8];
  const float* Wo        = (const float*)d_in[9];
  const float* bo        = (const float*)d_in[10];
  const float* ffn_ln_s  = (const float*)d_in[11];
  const float* ffn_ln_b  = (const float*)d_in[12];
  const float* W1        = (const float*)d_in[13];
  const float* b1        = (const float*)d_in[14];
  const float* W2        = (const float*)d_in[15];
  const float* b2        = (const float*)d_in[16];
  const float* last_ln_s = (const float*)d_in[17];
  const float* last_ln_b = (const float*)d_in[18];
  const float* iui_ln_s  = (const float*)d_in[19];
  const float* iui_ln_b  = (const float*)d_in[20];
  float* out = (float*)d_out;

  char* ws = (char*)d_ws;
  size_t off = 0;
  auto alloc = [&](size_t bytes) {
    size_t p = off;
    off = (off + bytes + 255) & ~(size_t)255;
    return p;
  };
  float*  S_f32 = (float*)(ws + alloc((size_t)M * D * 4));
  __bf16* S_bf  = (__bf16*)(ws + alloc((size_t)M * D * 2));
  float*  T_f32 = (float*)(ws + alloc((size_t)M * D * 4));
  __bf16* T_bf  = (__bf16*)(ws + alloc((size_t)M * D * 2));
  __bf16* Q_bf  = (__bf16*)(ws + alloc((size_t)M * D * 2));   // also FFN h
  __bf16* K_bf  = (__bf16*)(ws + alloc((size_t)M * D * 2));
  __bf16* VT_bf = (__bf16*)(ws + alloc((size_t)Bn * 2 * 64 * Lpad * 2));
  __bf16* A_bf  = (__bf16*)(ws + alloc((size_t)M * D * 2));
  __bf16* Wqkv_bf = (__bf16*)(ws + alloc((size_t)2 * 384 * 128 * 2));
  __bf16* Wo_bf   = (__bf16*)(ws + alloc((size_t)2 * 128 * 128 * 2));
  __bf16* W1_bf   = (__bf16*)(ws + alloc((size_t)2 * 128 * 128 * 2));
  __bf16* W2_bf   = (__bf16*)(ws + alloc((size_t)2 * 128 * 128 * 2));

  // weights -> bf16
  f32_to_bf16_kernel<<<384, 256, 0, stream>>>(Wqkv, Wqkv_bf, 2 * 384 * 128);
  f32_to_bf16_kernel<<<128, 256, 0, stream>>>(Wo, Wo_bf, 2 * 128 * 128);
  f32_to_bf16_kernel<<<128, 256, 0, stream>>>(W1, W1_bf, 2 * 128 * 128);
  f32_to_bf16_kernel<<<128, 256, 0, stream>>>(W2, W2_bf, 2 * 128 * 128);

  embed_kernel<<<M, 128, 0, stream>>>(logs, item_emb, pos_emb, S_f32, S_bf, L);

  const int tiles32 = M / 32;                 // 1600
  for (int i = 0; i < 2; ++i) {
    const __bf16* Wqkv_i = Wqkv_bf + (size_t)i * 384 * 128;
    // q = LN(seqs)
    ln_kernel<<<M, 128, 0, stream>>>(S_f32, attn_ln_s + i * D,
                                     attn_ln_b + i * D, T_f32, T_bf);
    // Q = q @ Wq^T + bq          (N=128 -> 2 strips)
    gemm_bf16_kernel<<<tiles32 * 2 / 4, 128, 0, stream>>>(
        T_bf, Wqkv_i, bqkv + i * 384, nullptr, nullptr, nullptr, Q_bf,
        M, 128, 128, 0);
    // K/V = seqs @ [Wk;Wv]^T + b (N=256 -> 4 strips), V stored transposed
    kv_gemm_kernel<<<tiles32 * 4 / 4, 128, 0, stream>>>(
        S_bf, Wqkv_i + 128 * 128, bqkv + i * 384 + 128, K_bf, VT_bf,
        M, 128, L, Lpad);
    // attention
    attn_kernel<<<Bn * 2 * 13, 32, 0, stream>>>(Q_bf, K_bf, VT_bf, A_bf, L);
    // seqs = q + attn @ Wo^T + bo
    gemm_bf16_kernel<<<tiles32 * 2 / 4, 128, 0, stream>>>(
        A_bf, Wo_bf + (size_t)i * 128 * 128, bo + i * D, T_f32, nullptr,
        S_f32, nullptr, M, 128, 128, 0);
    // x = LN(seqs)
    ln_kernel<<<M, 128, 0, stream>>>(S_f32, ffn_ln_s + i * D,
                                     ffn_ln_b + i * D, T_f32, T_bf);
    // h = relu(x @ W1^T + b1)
    gemm_bf16_kernel<<<tiles32 * 2 / 4, 128, 0, stream>>>(
        T_bf, W1_bf + (size_t)i * 128 * 128, b1 + i * D, nullptr, nullptr,
        nullptr, Q_bf, M, 128, 128, 1);
    // seqs = (h @ W2^T + b2 + x) * mask
    gemm_bf16_kernel<<<tiles32 * 2 / 4, 128, 0, stream>>>(
        Q_bf, W2_bf + (size_t)i * 128 * 128, b2 + i * D, T_f32, logs,
        S_f32, S_bf, M, 128, 128, 0);
  }
  // feats = LN(seqs) -> out
  ln_kernel<<<M, 128, 0, stream>>>(S_f32, last_ln_s, last_ln_b, out, nullptr);
  // out += neighbor head
  nbr_kernel<<<M, 128, 0, stream>>>(logs, nbr, item_emb, user_emb, iui_ln_s,
                                    iui_ln_b, out);
}
